// EnhancedGCN_70231305225145
// MI455X (gfx1250) — matrix operand order repaired and checked by reading, not compiled, not run on hardware
//
#include <hip/hip_runtime.h>

// EnhancedGCN forward for MI455X (gfx1250, wave32, WMMA).
// GEMMs: v_wmma_f32_16x16x32_bf16, A converted f32->bf16 on the fly
// (packed v_cvt_pk_bf16_f32 when available), weights pre-converted to bf16
// once and staged into LDS via GLOBAL_LOAD_ASYNC_TO_LDS_B128 when available.
// Graph aggregation: atomicAdd/atomicMax (HBM-bound, ~6GB edge traffic).
// Top-k threshold: device-side 4-pass radix select (graph-capture safe).

#ifndef __has_builtin
#define __has_builtin(x) 0
#endif

#if __has_builtin(__builtin_amdgcn_global_load_async_to_lds_b128) && \
    __has_builtin(__builtin_amdgcn_s_wait_asynccnt)
#define USE_ASYNC_LDS 1
#else
#define USE_ASYNC_LDS 0
#endif

#if __has_builtin(__builtin_amdgcn_cvt_pk_bf16_f32)
#define USE_PK_BF16 1
#else
#define USE_PK_BF16 0
#endif

typedef __attribute__((ext_vector_type(16))) __bf16 v16bf;
typedef __attribute__((ext_vector_type(2)))  __bf16 v2bf;
typedef __attribute__((ext_vector_type(8)))  float  v8f;
typedef int v4i_vec __attribute__((vector_size(16)));   // gcc-style, matches builtin param
#define AS1 __attribute__((address_space(1)))
#define AS3 __attribute__((address_space(3)))

#define EPSV 1e-5f

__device__ __forceinline__ float lrelu_f(float x){ return x >= 0.f ? x : 0.2f*x; }
__device__ __forceinline__ float sigmoid_f(float x){ return 1.f/(1.f + __expf(-x)); }

// round-to-nearest-even f32 -> bf16 (software fallback)
__device__ __forceinline__ unsigned short f2bf(float f){
  unsigned u = __float_as_uint(f);
  u += 0x7fffu + ((u >> 16) & 1u);
  return (unsigned short)(u >> 16);
}
// order-preserving float <-> uint transforms (for atomicMax / radix select)
__device__ __forceinline__ unsigned fmono(float f){
  unsigned u = __float_as_uint(f);
  return (u & 0x80000000u) ? ~u : (u | 0x80000000u);
}
__device__ __forceinline__ float imono(unsigned u){
  u = (u & 0x80000000u) ? (u & 0x7fffffffu) : ~u;
  return __uint_as_float(u);
}

// ---------------------------------------------------------------------------
// WMMA GEMM: C[M x Nout] = post( A[M x K] @ Wbf[Nout x K]^T + bias )
// A is f32 (converted to bf16 while staging); Wbf is pre-converted bf16.
// post: optional LeakyReLU then optional eval-BN (bnp = [gamma,beta,mean,var]).
// Block = 256 threads (8 waves). Tile BM=128 x BN=64 x BK=32.
// ---------------------------------------------------------------------------
#define BM 128
#define BN 64
#define BK 32
#define AST 56   // LDS row stride in halves (112B: 16B aligned, conflict-free)
#define BST 56

__global__ __launch_bounds__(256)
void k_gemm_wmma(const float* __restrict__ A, const unsigned short* __restrict__ Wbf,
                 const float* __restrict__ bias, const float* __restrict__ bnp,
                 float* __restrict__ C, long ldc, int M, int K, int Nout, int do_lrelu)
{
  __shared__ unsigned short As[BM*AST];
  __shared__ unsigned short Bs[BN*BST];
  const int tid  = threadIdx.x;
  const int lane = tid & 31, wv = tid >> 5;
  const int m0 = blockIdx.x * BM, n0 = blockIdx.y * BN;

  v8f acc[4] = {};

  const int arow = tid >> 1, acg = (tid & 1) * 16;   // A tile: 128 rows x 32 cols
  const int brow = tid >> 2, bcg = (tid & 3) * 8;    // B tile:  64 rows x 32 cols
  const int gm = m0 + arow;
  const float* aptr = (gm < M) ? (A + (size_t)gm * K + acg) : nullptr;
  const unsigned short* bptr = Wbf + (size_t)(n0 + brow) * K + bcg;
  unsigned short* asd = &As[arow*AST + acg];
  unsigned short* bsd = &Bs[brow*BST + bcg];

  for (int k0 = 0; k0 < K; k0 += BK) {
    // ---- stage B tile (bf16 weights, 16B per thread) ----
#if USE_ASYNC_LDS
    __builtin_amdgcn_global_load_async_to_lds_b128(
        (AS1 v4i_vec*)(bptr + k0),
        (AS3 v4i_vec*)bsd, 0, 0);
#else
    *(uint4*)bsd = *(const uint4*)(bptr + k0);
#endif
    // ---- stage A tile (f32 -> bf16 convert) ----
    if (aptr) {
      if (k0 + BK < K) __builtin_prefetch(aptr + k0 + BK, 0, 1);
#if USE_PK_BF16
      #pragma unroll
      for (int j = 0; j < 16; j += 2)
        *(v2bf*)&asd[j] = __builtin_amdgcn_cvt_pk_bf16_f32(aptr[k0 + j], aptr[k0 + j + 1]);
#else
      #pragma unroll
      for (int j = 0; j < 16; ++j) asd[j] = f2bf(aptr[k0 + j]);
#endif
    } else {
      #pragma unroll
      for (int j = 0; j < 16; ++j) asd[j] = 0;
    }
#if USE_ASYNC_LDS
    __builtin_amdgcn_s_wait_asynccnt(0);
#endif
    __syncthreads();

    // A fragment (16x32 bf16): lane<16: M=lane, halves = K[0..7],K[16..23]
    //                          lane>=16: M=lane-16, halves = K[8..15],K[24..31]
    v16bf afrag;
    {
      const unsigned short* p = &As[(wv*16 + (lane & 15))*AST + ((lane < 16) ? 0 : 8)];
      ((uint4*)&afrag)[0] = *(const uint4*)(p);
      ((uint4*)&afrag)[1] = *(const uint4*)(p + 16);
    }
    #pragma unroll
    for (int nt = 0; nt < 4; ++nt) {
      // B fragment (32x16 bf16): lane<16: N=lane, halves = K[0..7],K[16..23]
      //                          lane>=16: N=lane-16, halves = K[8..15],K[24..31]
      v16bf bfrag;
      const unsigned short* p = &Bs[(nt*16 + (lane & 15))*BST + ((lane < 16) ? 0 : 8)];
      ((uint4*)&bfrag)[0] = *(const uint4*)(p);
      ((uint4*)&bfrag)[1] = *(const uint4*)(p + 16);
      acc[nt] = __builtin_amdgcn_wmma_f32_16x16x32_bf16(false, afrag, false, bfrag,
                                                        (short)0, acc[nt], false, false);
    }
    __syncthreads();
  }

  // D layout: lane<16 -> N=lane,    VGPR r -> M = r
  //           lane>=16 -> N=lane-16, VGPR r -> M = r+8
  const int mrow = m0 + wv*16 + ((lane < 16) ? 0 : 8);
  const int ncol = lane & 15;
  #pragma unroll
  for (int nt = 0; nt < 4; ++nt) {
    const int n = n0 + nt*16 + ncol;
    const float bv = bias ? bias[n] : 0.f;
    float g = 0.f, bb = 0.f, mu = 0.f, rs = 0.f;
    if (bnp) { g = bnp[n]; bb = bnp[Nout+n]; mu = bnp[2*Nout+n]; rs = rsqrtf(bnp[3*Nout+n] + EPSV); }
    #pragma unroll
    for (int r = 0; r < 8; ++r) {
      const int m = mrow + r;
      if (m < M) {
        float v = acc[nt][r] + bv;
        if (do_lrelu) v = lrelu_f(v);
        if (bnp) v = g*(v - mu)*rs + bb;
        C[(size_t)m*ldc + n] = v;
      }
    }
  }
}

// one-time weight conversion f32 -> bf16
__global__ void k_f32_to_bf16(const float* __restrict__ src, unsigned short* __restrict__ dst, long n){
  long i = (long)blockIdx.x*256 + threadIdx.x;
  if (i < n) dst[i] = f2bf(src[i]);
}

// ---------------------------------------------------------------------------
// Feature fusion: wave-per-node, 8 nodes per 256-thread block, LDS staging.
// ---------------------------------------------------------------------------
__global__ __launch_bounds__(256)
void k_fusion(const float* __restrict__ x,
              const float* cen_W, const float* cen_b, const float* cen_bn,
              const float* emb_W, const float* emb_b, const float* emb_bn,
              const float* fus_W, const float* fus_b,
              const float* fa_W1, const float* fa_b1,
              const float* fa_W2, const float* fa_b2,
              const float* adj_W, const float* adj_b,
              float* __restrict__ state, int N)
{
  __shared__ float sm[8*424];
  const int lane = threadIdx.x & 31, wv = threadIdx.x >> 5;
  const int node = blockIdx.x*8 + wv;
  const bool act = node < N;
  float* xrow = sm + wv*424;     // 135 (+1 pad)
  float* ce   = xrow + 136;      // 96  (c then e, concat order)
  float* hbuf = ce + 96;         // 64
  float* tbuf = hbuf + 64;       // 64
  float* hhb  = tbuf + 64;       // 64

  if (act) for (int j = lane; j < 135; j += 32) xrow[j] = x[(size_t)node*135 + j];
  __syncthreads();
  if (act) {
    { // c: 32 channels, 7 inputs
      float a = cen_b[lane];
      #pragma unroll
      for (int k = 0; k < 7; ++k) a += xrow[k]*cen_W[lane*7+k];
      a = lrelu_f(a);
      a = cen_bn[lane]*(a - cen_bn[64+lane])*rsqrtf(cen_bn[96+lane]+EPSV) + cen_bn[32+lane];
      ce[lane] = a;
    }
    for (int jj = 0; jj < 2; ++jj) { // e: 64 channels, 128 inputs
      int j = lane + jj*32;
      float a = emb_b[j];
      for (int k = 0; k < 128; ++k) a += xrow[7+k]*emb_W[j*128+k];
      a = lrelu_f(a);
      a = emb_bn[j]*(a - emb_bn[128+j])*rsqrtf(emb_bn[192+j]+EPSV) + emb_bn[64+j];
      ce[32+j] = a;
    }
  }
  __syncthreads();
  if (act) for (int jj = 0; jj < 2; ++jj) {   // h = [c,e] @ fus_W^T + b
    int j = lane + jj*32;
    float a = fus_b[j];
    for (int k = 0; k < 96; ++k) a += ce[k]*fus_W[j*96+k];
    hbuf[j] = a;
  }
  __syncthreads();
  if (act) for (int jj = 0; jj < 2; ++jj) {   // t = lrelu(h @ fa_W1^T + b)
    int j = lane + jj*32;
    float a = fa_b1[j];
    for (int k = 0; k < 64; ++k) a += hbuf[k]*fa_W1[j*64+k];
    tbuf[j] = lrelu_f(a);
  }
  __syncthreads();
  if (act) for (int jj = 0; jj < 2; ++jj) {   // hh = h * sigmoid(t @ fa_W2^T + b)
    int j = lane + jj*32;
    float a = fa_b2[j];
    for (int k = 0; k < 64; ++k) a += tbuf[k]*fa_W2[j*64+k];
    hhb[j] = hbuf[j]*sigmoid_f(a);
  }
  __syncthreads();
  if (act) for (int jj = 0; jj < 8; ++jj) {   // state = hh @ adj_W^T + b  (64->256)
    int j = lane + jj*32;
    float a = adj_b[j];
    for (int k = 0; k < 64; ++k) a += hhb[k]*adj_W[j*64+k];
    state[(size_t)node*256 + j] = a;
  }
}

// ---------------------------------------------------------------------------
// Graph kernels
// ---------------------------------------------------------------------------
__global__ void k_deg_init(float* deg, int N){
  int n = blockIdx.x*256 + threadIdx.x; if (n < N) deg[n] = 1.f;  // self-loop
}
__global__ void k_deg_edge(const int* __restrict__ dst, float* deg, int E){
  int e = blockIdx.x*256 + threadIdx.x; if (e < E) atomicAdd(&deg[dst[e]], 1.f);
}
__global__ void k_rsqrt_ip(float* d, int N){
  int n = blockIdx.x*256 + threadIdx.x; if (n < N) d[n] = rsqrtf(d[n]);
}
// GCN: out = bias + self-loop term, then atomic scatter of edges
__global__ void k_gcn_init(const float* __restrict__ g, const float* __restrict__ dinv,
                           const float* __restrict__ bias, float* __restrict__ out, int N){
  int i = blockIdx.x*256 + threadIdx.x;
  if (i < N*256){ int n = i >> 8, c = i & 255; float di = dinv[n]; out[i] = bias[c] + g[i]*di*di; }
}
__global__ void k_gcn_edge(const int* __restrict__ src, const int* __restrict__ dst,
                           const float* __restrict__ g, const float* __restrict__ dinv,
                           float* __restrict__ out, int E){
  long gid = (long)blockIdx.x*256 + threadIdx.x;
  if (gid >= (long)E*64) return;
  int e = (int)(gid >> 6), c = (int)(gid & 63) << 2;
  int s = src[e], d = dst[e];
  float nrm = dinv[s]*dinv[d];
  const float4 gv = *(const float4*)(g + (size_t)s*256 + c);
  float* o = out + (size_t)d*256 + c;
  atomicAdd(o+0, gv.x*nrm); atomicAdd(o+1, gv.y*nrm);
  atomicAdd(o+2, gv.z*nrm); atomicAdd(o+3, gv.w*nrm);
}
// GAT pieces
__global__ void k_asad(const float* __restrict__ hh, const float* asrc, const float* adst,
                       float* a_s, float* a_d, int N){
  int i = blockIdx.x*256 + threadIdx.x; if (i >= N*8) return;
  int n = i >> 3, h = i & 7;
  const float* hp = hh + (size_t)n*256 + h*32;
  float s = 0.f, d = 0.f;
  #pragma unroll
  for (int k = 0; k < 32; ++k){ s += hp[k]*asrc[h*32+k]; d += hp[k]*adst[h*32+k]; }
  a_s[i] = s; a_d[i] = d;
}
__global__ void k_mx_init(const float* a_s, const float* a_d, unsigned* mxu, int N){
  int i = blockIdx.x*256 + threadIdx.x; if (i >= N*8) return;
  mxu[i] = fmono(lrelu_f(a_s[i] + a_d[i]));   // self-loop logit seeds the max
}
__global__ void k_edge_max(const int* src, const int* dst, const float* a_s,
                           const float* a_d, unsigned* mxu, int E){
  long gid = (long)blockIdx.x*256 + threadIdx.x;
  if (gid >= (long)E*8) return;
  int e = (int)(gid >> 3), h = (int)(gid & 7);
  int s = src[e], d = dst[e];
  float l = lrelu_f(a_s[s*8+h] + a_d[d*8+h]);
  atomicMax(&mxu[d*8+h], fmono(l));
}
__global__ void k_den_init(const float* a_s, const float* a_d, const unsigned* mxu,
                           float* den, float* exSelf, int N){
  int i = blockIdx.x*256 + threadIdx.x; if (i >= N*8) return;
  float es = __expf(lrelu_f(a_s[i] + a_d[i]) - imono(mxu[i]));
  exSelf[i] = es; den[i] = es;
}
__global__ void k_edge_ex(const int* src, const int* dst, const float* a_s, const float* a_d,
                          const unsigned* mxu, float* den, float* exE, int E){
  long gid = (long)blockIdx.x*256 + threadIdx.x;
  if (gid >= (long)E*8) return;
  int e = (int)(gid >> 3), h = (int)(gid & 7);
  int s = src[e], d = dst[e];
  float ex = __expf(lrelu_f(a_s[s*8+h] + a_d[d*8+h]) - imono(mxu[d*8+h]));
  exE[gid] = ex; atomicAdd(&den[d*8+h], ex);
}
__global__ void k_gat_init(const float* __restrict__ hh, const float* exSelf, const float* den,
                           const float* bias, float* __restrict__ out, int N){
  int i = blockIdx.x*256 + threadIdx.x; if (i >= N*256) return;
  int n = i >> 8, c = i & 255, h = c >> 5;
  out[i] = bias[c] + (exSelf[n*8+h]/den[n*8+h])*hh[i];
}
__global__ void k_gat_edge(const int* __restrict__ src, const int* __restrict__ dst,
                           const float* __restrict__ hh, const float* __restrict__ exE,
                           const float* __restrict__ den, float* __restrict__ out, int E){
  long gid = (long)blockIdx.x*256 + threadIdx.x;
  if (gid >= (long)E*64) return;
  int e = (int)(gid >> 6), c = (int)(gid & 63) << 2, h = c >> 5;
  int s = src[e], d = dst[e];
  float al = exE[(size_t)e*8 + h]/den[d*8+h];
  const float4 hv = *(const float4*)(hh + (size_t)s*256 + c);
  float* o = out + (size_t)d*256 + c;
  atomicAdd(o+0, hv.x*al); atomicAdd(o+1, hv.y*al);
  atomicAdd(o+2, hv.z*al); atomicAdd(o+3, hv.w*al);
}
// BN -> lrelu -> sigmoid gate -> residual; writes new state + jk slice
__global__ void k_post(const float* __restrict__ gin, const float* __restrict__ bnp,
                       const float* __restrict__ gate_W, const float* gate_b_all,
                       float* __restrict__ state, float* __restrict__ jk, int layer, int N){
  int n = blockIdx.x*256 + threadIdx.x; if (n >= N) return;
  const float* gp = gin + (size_t)n*256;
  float gd = gate_b_all[layer];
  for (int c = 0; c < 256; ++c){
    float v = bnp[c]*(gp[c] - bnp[512+c])*rsqrtf(bnp[768+c]+EPSV) + bnp[256+c];
    v = lrelu_f(v);
    gd += v*gate_W[c];
  }
  float gate = sigmoid_f(gd);
  float* sp = state + (size_t)n*256;
  float* jp = jk + (size_t)n*1024 + (size_t)layer*256;
  for (int c = 0; c < 256; ++c){
    float v = bnp[c]*(gp[c] - bnp[512+c])*rsqrtf(bnp[768+c]+EPSV) + bnp[256+c];
    v = lrelu_f(v);
    float ns = gate*v + (1.f - gate)*sp[c];
    sp[c] = ns; jp[c] = ns;
  }
}
// final scalar head + scaling by 1/temperature
__global__ void k_mlp3(const float* __restrict__ m2, const float* W3, const float* b3,
                       const float* temp, float* scaled, int N){
  int n = blockIdx.x*256 + threadIdx.x; if (n >= N) return;
  const float* p = m2 + (size_t)n*256;
  float a = b3[0];
  for (int c = 0; c < 256; ++c) a += p[c]*W3[c];
  scaled[n] = sigmoid_f(a)/temp[0];
}
// ---------------------------------------------------------------------------
// Softmax over N + top-k threshold (radix select) + boost + cube
// red[0]=max(mono) red[1]=sum(float) red[2]=prefix red[3]=rank
// ---------------------------------------------------------------------------
__global__ void k_red_init(unsigned* red, int topk){
  red[0] = 0u; ((float*)red)[1] = 0.f; red[2] = 0u; red[3] = (unsigned)topk;
}
__global__ void k_max(const float* s, unsigned* red, int N){
  int n = blockIdx.x*256 + threadIdx.x; if (n < N) atomicMax(&red[0], fmono(s[n]));
}
__global__ void k_expsum(const float* s, const unsigned* red, float* tmp, float* sum, int N){
  int n = blockIdx.x*256 + threadIdx.x; if (n >= N) return;
  float t = __expf(s[n] - imono(red[0])); tmp[n] = t; atomicAdd(sum, t);
}
__global__ void k_zero_bins(unsigned* bins){ bins[threadIdx.x] = 0u; }
__global__ void k_count(const float* s, const unsigned* red, unsigned* bins,
                        unsigned maskhigh, int shift, int N){
  int n = blockIdx.x*256 + threadIdx.x; if (n >= N) return;
  unsigned u = fmono(s[n]);
  if ((u & maskhigh) == (red[2] & maskhigh)) atomicAdd(&bins[(u >> shift) & 0xffu], 1u);
}
__global__ void k_select(unsigned* red, const unsigned* bins, int shift){
  if (threadIdx.x != 0) return;
  unsigned rank = red[3], cum = 0; int chosen = 0;
  for (int b = 255; b >= 0; --b){
    unsigned cb = bins[b];
    if (cum + cb >= rank){ chosen = b; break; }
    cum += cb;
  }
  red[2] |= ((unsigned)chosen) << shift;
  red[3] = rank - cum;
}
__global__ void k_final(const float* s, const float* tmp, const unsigned* red, float* out, int N){
  int n = blockIdx.x*256 + threadIdx.x; if (n >= N) return;
  float enh = tmp[n]/((const float*)red)[1];
  float mk = (fmono(s[n]) >= red[2]) ? 1.f : 0.1f;
  float b = enh*mk;
  out[n] = b*b*b;
}

// ---------------------------------------------------------------------------
extern "C" void kernel_launch(void* const* d_in, const int* in_sizes, int n_in,
                              void* d_out, int out_size, void* d_ws, size_t ws_size,
                              hipStream_t stream)
{
  const float* x   = (const float*)d_in[0];
  const int*   ei  = (const int*)  d_in[1];
  const int N = in_sizes[0]/135;
  const int E = in_sizes[1]/2;
  const int* src = ei;
  const int* dst = ei + E;
  const float* cen_W=(const float*)d_in[2];  const float* cen_b=(const float*)d_in[3];
  const float* cen_bn=(const float*)d_in[4]; const float* emb_W=(const float*)d_in[5];
  const float* emb_b=(const float*)d_in[6];  const float* emb_bn=(const float*)d_in[7];
  const float* fus_W=(const float*)d_in[8];  const float* fus_b=(const float*)d_in[9];
  const float* fa_W1=(const float*)d_in[10]; const float* fa_b1=(const float*)d_in[11];
  const float* fa_W2=(const float*)d_in[12]; const float* fa_b2=(const float*)d_in[13];
  const float* adj_W=(const float*)d_in[14]; const float* adj_b=(const float*)d_in[15];
  const float* gcn_W=(const float*)d_in[16]; const float* gcn_b=(const float*)d_in[17];
  const float* gat_W=(const float*)d_in[18]; const float* gat_b=(const float*)d_in[19];
  const float* gat_asrc=(const float*)d_in[20]; const float* gat_adst=(const float*)d_in[21];
  const float* layer_bn=(const float*)d_in[22];
  const float* gate_W=(const float*)d_in[23]; const float* gate_b=(const float*)d_in[24];
  const float* lin1_W=(const float*)d_in[25]; const float* lin1_b=(const float*)d_in[26];
  const float* mlp_W1=(const float*)d_in[27]; const float* mlp_b1=(const float*)d_in[28];
  const float* mlp_bn1=(const float*)d_in[29];const float* mlp_W2=(const float*)d_in[30];
  const float* mlp_b2=(const float*)d_in[31]; const float* mlp_bn2=(const float*)d_in[32];
  const float* mlp_W3=(const float*)d_in[33]; const float* mlp_b3=(const float*)d_in[34];
  const float* temperature=(const float*)d_in[35];
  (void)n_in; (void)out_size; (void)ws_size;

  // workspace layout (floats)
  float* ws = (float*)d_ws; size_t o = 0;
  float* state  = ws + o; o += (size_t)N*256;
  float* bufG   = ws + o; o += (size_t)N*256;   // gemm outputs (g, hh, h2)
  float* bufCur = ws + o; o += (size_t)N*256;   // post-GCN activations / m2
  float* bufOut = ws + o; o += (size_t)N*256;   // GAT aggregation output
  float* jk     = ws + o; o += (size_t)N*1024;  // JK concat (aliased by m1 [N,512])
  float* dinv   = ws + o; o += (size_t)N;
  float* a_s    = ws + o; o += (size_t)N*8;
  float* a_d    = ws + o; o += (size_t)N*8;
  unsigned* mxu = (unsigned*)(ws + o); o += (size_t)N*8;
  float* den    = ws + o; o += (size_t)N*8;
  float* exSelf = ws + o; o += (size_t)N*8;
  float* exE    = ws + o; o += (size_t)E*8;
  float* scaled = ws + o; o += (size_t)N;
  float* stmp   = ws + o; o += (size_t)N;
  unsigned* red  = (unsigned*)(ws + o); o += 8;
  unsigned* bins = (unsigned*)(ws + o); o += 256;
  // bf16 weight cache (1,048,576 halves = 524,288 floats)
  unsigned short* wbf = (unsigned short*)(ws + o); o += 524288;
  unsigned short* wb_gcn  = wbf;            // 4 x 256*256
  unsigned short* wb_gat  = wbf + 262144;   // 4 x 256*256
  unsigned short* wb_lin1 = wbf + 524288;   // 256*1024
  unsigned short* wb_W1   = wbf + 786432;   // 512*256
  unsigned short* wb_W2   = wbf + 917504;   // 256*512

  auto cdiv = [](long a, long b){ return (int)((a + b - 1)/b); };

  // 0) one-time weight conversion to bf16
  k_f32_to_bf16<<<cdiv(262144,256),256,0,stream>>>(gcn_W,  wb_gcn,  262144);
  k_f32_to_bf16<<<cdiv(262144,256),256,0,stream>>>(gat_W,  wb_gat,  262144);
  k_f32_to_bf16<<<cdiv(262144,256),256,0,stream>>>(lin1_W, wb_lin1, 262144);
  k_f32_to_bf16<<<cdiv(131072,256),256,0,stream>>>(mlp_W1, wb_W1,   131072);
  k_f32_to_bf16<<<cdiv(131072,256),256,0,stream>>>(mlp_W2, wb_W2,   131072);

  // 1) feature fusion -> state [N,256]
  k_fusion<<<cdiv(N,8),256,0,stream>>>(x, cen_W,cen_b,cen_bn, emb_W,emb_b,emb_bn,
                                       fus_W,fus_b, fa_W1,fa_b1, fa_W2,fa_b2,
                                       adj_W,adj_b, state, N);
  // 2) degrees with self-loops -> dinv
  k_deg_init<<<cdiv(N,256),256,0,stream>>>(dinv, N);
  k_deg_edge<<<cdiv(E,256),256,0,stream>>>(dst, dinv, E);
  k_rsqrt_ip<<<cdiv(N,256),256,0,stream>>>(dinv, N);

  const dim3 g256(cdiv(N,BM), 256/BN);
  // 3) GNN layers
  for (int l = 0; l < 4; ++l) {
    k_gemm_wmma<<<g256,256,0,stream>>>(state, wb_gcn + (size_t)l*65536,
                                       nullptr, nullptr, bufG, 256, N, 256, 256, 0);
    k_gcn_init<<<cdiv((long)N*256,256),256,0,stream>>>(bufG, dinv, gcn_b + l*256, bufCur, N);
    k_gcn_edge<<<cdiv((long)E*64,256),256,0,stream>>>(src, dst, bufG, dinv, bufCur, E);

    k_gemm_wmma<<<g256,256,0,stream>>>(bufCur, wb_gat + (size_t)l*65536,
                                       nullptr, nullptr, bufG, 256, N, 256, 256, 0);
    k_asad<<<cdiv((long)N*8,256),256,0,stream>>>(bufG, gat_asrc + l*256, gat_adst + l*256,
                                                 a_s, a_d, N);
    k_mx_init <<<cdiv((long)N*8,256),256,0,stream>>>(a_s, a_d, mxu, N);
    k_edge_max<<<cdiv((long)E*8,256),256,0,stream>>>(src, dst, a_s, a_d, mxu, E);
    k_den_init<<<cdiv((long)N*8,256),256,0,stream>>>(a_s, a_d, mxu, den, exSelf, N);
    k_edge_ex <<<cdiv((long)E*8,256),256,0,stream>>>(src, dst, a_s, a_d, mxu, den, exE, E);
    k_gat_init<<<cdiv((long)N*256,256),256,0,stream>>>(bufG, exSelf, den, gat_b + l*256, bufOut, N);
    k_gat_edge<<<cdiv((long)E*64,256),256,0,stream>>>(src, dst, bufG, exE, den, bufOut, E);
    k_post<<<cdiv(N,256),256,0,stream>>>(bufOut, layer_bn + (size_t)l*1024,
                                         gate_W + l*256, gate_b, state, jk, l, N);
  }
  // 4) head: lin1 (K=1024) -> mlp1 (512) -> mlp2 (256) -> scalar
  k_gemm_wmma<<<dim3(cdiv(N,BM),256/BN),256,0,stream>>>(jk,   wb_lin1, lin1_b, nullptr,
                                                        bufG, 256, N, 1024, 256, 1);
  k_gemm_wmma<<<dim3(cdiv(N,BM),512/BN),256,0,stream>>>(bufG, wb_W1,  mlp_b1, mlp_bn1,
                                                        jk,   512, N, 256, 512, 1);
  k_gemm_wmma<<<dim3(cdiv(N,BM),256/BN),256,0,stream>>>(jk,   wb_W2,  mlp_b2, mlp_bn2,
                                                        bufCur, 256, N, 512, 256, 1);
  k_mlp3<<<cdiv(N,256),256,0,stream>>>(bufCur, mlp_W3, mlp_b3, temperature, scaled, N);

  // 5) softmax over N + radix-select threshold + boost + cube
  int topk = (int)(N*0.2); if (topk < 1) topk = 1;
  k_red_init<<<1,1,0,stream>>>(red, topk);
  k_max   <<<cdiv(N,256),256,0,stream>>>(scaled, red, N);
  k_expsum<<<cdiv(N,256),256,0,stream>>>(scaled, red, stmp, ((float*)red)+1, N);
  for (int p = 0; p < 4; ++p) {
    int shift = 24 - 8*p;
    unsigned maskhigh = (p == 0) ? 0u : (0xffffffffu << (32 - 8*p));
    k_zero_bins<<<1,256,0,stream>>>(bins);
    k_count<<<cdiv(N,256),256,0,stream>>>(scaled, red, bins, maskhigh, shift, N);
    k_select<<<1,1,0,stream>>>(red, bins, shift);
  }
  k_final<<<cdiv(N,256),256,0,stream>>>(scaled, stmp, red, (float*)d_out, N);
}
